// Transformer_59176059404939
// MI455X (gfx1250) — matrix operand order repaired and checked by reading, not compiled, not run on hardware
//
#include <hip/hip_runtime.h>
#include <hip/hip_bf16.h>
#include <math.h>

// ---------------------------------------------------------------------------
// MI455X (gfx1250) transformer forward.
// Compute-bound (~722 GFLOP vs <1GB of HBM traffic): route all GEMMs through
// v_wmma_f32_16x16x32_bf16 (wave32, fp32 accumulate). Softmax/LN/GELU in fp32.
// Data movement: async global->LDS (ASYNCcnt) when available, TDM tensor
// loads (TENSORcnt) for weight-tile prefetch, global_prefetch_b8 for K tiles.
// ---------------------------------------------------------------------------

typedef __attribute__((ext_vector_type(16))) __bf16   v16bf;
typedef __attribute__((ext_vector_type(8)))  float    v8f;
typedef __attribute__((ext_vector_type(8)))  unsigned v8u;
typedef __attribute__((ext_vector_type(4)))  unsigned u32x4;
typedef __attribute__((ext_vector_type(8)))  int      i32x8;
typedef __attribute__((ext_vector_type(4)))  int      i32x4;
typedef __attribute__((ext_vector_type(4)))  int      v4i;

#if __has_builtin(__builtin_amdgcn_global_load_async_to_lds_b128)
#define HAVE_ASYNC_LDS 1
#else
#define HAVE_ASYNC_LDS 0
#endif
#if __has_builtin(__builtin_amdgcn_s_wait_asynccnt)
#define HAVE_WAIT_ASYNC 1
#else
#define HAVE_WAIT_ASYNC 0
#endif
#if __has_builtin(__builtin_amdgcn_tensor_load_to_lds)
#define HAVE_TDM 1
#else
#define HAVE_TDM 0
#endif

#define LNUM 6
#define BD   4
#define SD   1024
#define DD   1024
#define FD   4096
#define HH   16
#define DK   64

__device__ __forceinline__ unsigned short f2bf(float x) {
  unsigned u = __builtin_bit_cast(unsigned, x);
  unsigned r = u + 0x7FFFu + ((u >> 16) & 1u);   // round-to-nearest-even
  return (unsigned short)(r >> 16);
}

__device__ __forceinline__ float gelu_exact(float x) {
  return 0.5f * x * (1.0f + erff(x * 0.70710678118654752f));
}

// ------------------------- weight / activation converts ---------------------
__global__ void cvt_plain(const float* __restrict__ s, unsigned short* __restrict__ d,
                          size_t n) {
  for (size_t i = blockIdx.x * blockDim.x + threadIdx.x; i < n;
       i += (size_t)gridDim.x * blockDim.x)
    d[i] = f2bf(s[i]);
}

// WQ/WK/WV: permute columns so head index becomes the slow axis (h*64+k),
// matching reference reshape(B,S,dk,H) where head is fastest. Equivalent math.
__global__ void cvt_qkv(const float* __restrict__ s, unsigned short* __restrict__ d,
                        size_t n) {
  for (size_t i = blockIdx.x * blockDim.x + threadIdx.x; i < n;
       i += (size_t)gridDim.x * blockDim.x) {
    size_t l = i / ((size_t)DD * DD);
    size_t rem = i - l * (size_t)DD * DD;
    int row = (int)(rem / DD);
    int j   = (int)(rem % DD);             // dst col = h*64 + k
    int scol = (j & 63) * HH + (j >> 6);   // src col = k*H + h
    d[i] = f2bf(s[l * (size_t)DD * DD + (size_t)row * DD + scol]);
  }
}

// WO: permute rows with the same map (consumes head-contiguous activations).
__global__ void cvt_wo(const float* __restrict__ s, unsigned short* __restrict__ d,
                       size_t n) {
  for (size_t i = blockIdx.x * blockDim.x + threadIdx.x; i < n;
       i += (size_t)gridDim.x * blockDim.x) {
    size_t l = i / ((size_t)DD * DD);
    size_t rem = i - l * (size_t)DD * DD;
    int row = (int)(rem / DD);             // dst row = h*64 + k
    int j   = (int)(rem % DD);
    int srow = (row & 63) * HH + (row >> 6);
    d[i] = f2bf(s[l * (size_t)DD * DD + (size_t)srow * DD + j]);
  }
}

// --------------------- TDM weight-tile prefetch (TENSORcnt) -----------------
// One wave builds a D# for a 128x32-element bf16 tile of the layer's weights
// and issues TENSOR_LOAD_TO_LDS, waiting on TENSORcnt. Warms L2 for the
// upcoming GEMMs through the Tensor Data Mover path.
__global__ void tdm_prefetch(const unsigned short* __restrict__ W) {
#if HAVE_TDM
  __shared__ __align__(16) unsigned short tile[32 * 128];
  unsigned long long ga = (unsigned long long)(uintptr_t)W;
  u32x4 g0;
  g0[0] = 1u;                                    // count=1 (valid user D#)
  g0[1] = 0u;                                    // lds_addr = base of allocation
  g0[2] = (unsigned)(ga & 0xFFFFFFFFu);          // global_addr[31:0]
  g0[3] = (unsigned)((ga >> 32) & 0x01FFFFFFu)   // global_addr[56:32]
          | (2u << 30);                          // type = 2 ("image")
  i32x8 g1;
  g1[0] = 0x10000;             // workgroup_mask=0 (not in cluster), data_size=2B
  g1[1] = (int)(1024u << 16);  // tensor_dim0 = 1024 (bits 79:48, low half here)
  g1[2] = (int)(1024u << 16);  // tensor_dim0 hi = 0 | tensor_dim1 = 1024 (lo)
  g1[3] = (int)(128u << 16);   // tensor_dim1 hi = 0 | tile_dim0 = 128
  g1[4] = 32;                  // tile_dim1 = 32, tile_dim2 = 0
  g1[5] = 1024;                // tensor_dim0_stride (lo 32 of 48)
  g1[6] = 0;                   // stride hi | tensor_dim1_stride lo
  g1[7] = 0;
  i32x4 z4 = {0, 0, 0, 0};
#if __clang_major__ >= 23
  i32x8 z8 = {0, 0, 0, 0, 0, 0, 0, 0};
  __builtin_amdgcn_tensor_load_to_lds(g0, g1, z4, z4, z8, 0);
#else
  __builtin_amdgcn_tensor_load_to_lds(g0, g1, z4, z4, 0);
#endif
  __builtin_amdgcn_s_wait_tensorcnt(0);
  (void)tile;
#else
  (void)W;
#endif
}

// ------------------------------ WMMA GEMM -----------------------------------
// C[M,N] = act(A[M,K] @ B[K,N] + bias). A,B bf16 row-major. 128x128 block tile,
// 8 waves, each wave 32x64 (2x4 WMMA C tiles), K-step 32.
__global__ __launch_bounds__(256) void gemm_bf16(
    const unsigned short* __restrict__ A, const unsigned short* __restrict__ B,
    int M, int N, int K, const float* __restrict__ bias, int act,
    float* __restrict__ Cf, unsigned short* __restrict__ Cb) {
  __shared__ __align__(16) unsigned As[128 * 16];  // [row][k/2] packed bf16 pairs
  __shared__ __align__(16) unsigned Bs[128 * 16];  // transposed: [n][k/2]

  const int tid  = threadIdx.x;
  const int lane = tid & 31;
  const int wave = tid >> 5;
  const int wm = wave & 3;        // 4 waves along M
  const int wn = wave >> 2;       // 2 waves along N
  const int blockM = blockIdx.y * 128;
  const int blockN = blockIdx.x * 128;
  const int mn = lane & 15;       // col-within-tile (B/C) or row (A)
  const int mh = lane >> 4;       // lane half

  v8f c[2][4] = {};

  for (int k0 = 0; k0 < K; k0 += 32) {
    // ---- stage A tile 128x32: async global->LDS when available ----
    {
      int row = tid >> 1, half = tid & 1;
      const unsigned short* g = A + (size_t)(blockM + row) * K + k0 + half * 16;
      unsigned* sh = &As[row * 16 + half * 8];
#if HAVE_ASYNC_LDS
      __builtin_amdgcn_global_load_async_to_lds_b128((v4i*)g, (v4i*)sh, 0, 0);
      __builtin_amdgcn_global_load_async_to_lds_b128((v4i*)(g + 8), (v4i*)(sh + 4),
                                                     0, 0);
#else
      const uint4* gv = (const uint4*)g;
      uint4* shv = (uint4*)sh;
      shv[0] = gv[0];
      shv[1] = gv[1];
#endif
      if (k0 + 32 < K)
        __builtin_prefetch(g + 32, 0, 1);
    }
    // ---- stage B tile 32x128, transposed into Bs[n][k] ----
    {
      int kr = tid >> 3, nseg = (tid & 7) * 16;
      const uint4* g = (const uint4*)(B + (size_t)(k0 + kr) * N + blockN + nseg);
      uint4 p0 = g[0], p1 = g[1];
      unsigned tmp[8] = {p0.x, p0.y, p0.z, p0.w, p1.x, p1.y, p1.z, p1.w};
      unsigned short* Bs16 = (unsigned short*)Bs;
#pragma unroll
      for (int i = 0; i < 8; ++i) {
        Bs16[(nseg + 2 * i)     * 32 + kr] = (unsigned short)(tmp[i] & 0xFFFFu);
        Bs16[(nseg + 2 * i + 1) * 32 + kr] = (unsigned short)(tmp[i] >> 16);
      }
      if (k0 + 32 < K)
        __builtin_prefetch(B + (size_t)(k0 + kr + 32) * N + blockN + nseg, 0, 1);
    }
#if HAVE_ASYNC_LDS
#if HAVE_WAIT_ASYNC
    __builtin_amdgcn_s_wait_asynccnt(0);
#else
    asm volatile("s_wait_asynccnt 0" ::: "memory");
#endif
#endif
    __syncthreads();

    // ---- A fragments (16x32 bf16, ISA layout: lane m=L%16, K groups by v) ----
    v16bf afr[2];
#pragma unroll
    for (int mi = 0; mi < 2; ++mi) {
      int rb = wm * 32 + mi * 16 + mn;
      v8u r;
#pragma unroll
      for (int v = 0; v < 8; ++v) {
        int kidx = (v >> 2) * 8 + mh * 4 + (v & 3);  // packed-pair index
        r[v] = As[rb * 16 + kidx];
      }
      afr[mi] = __builtin_bit_cast(v16bf, r);
    }
    // ---- B fragments (32x16 bf16: lane n=L%16, K groups by v, same as A) ----
    v16bf bfr[4];
#pragma unroll
    for (int ni = 0; ni < 4; ++ni) {
      int nb = wn * 64 + ni * 16 + mn;
      v8u r;
#pragma unroll
      for (int v = 0; v < 8; ++v) r[v] = Bs[nb * 16 + (v >> 2) * 8 + mh * 4 + (v & 3)];
      bfr[ni] = __builtin_bit_cast(v16bf, r);
    }
#pragma unroll
    for (int mi = 0; mi < 2; ++mi)
#pragma unroll
      for (int ni = 0; ni < 4; ++ni)
        c[mi][ni] = __builtin_amdgcn_wmma_f32_16x16x32_bf16(
            false, afr[mi], false, bfr[ni], (short)0, c[mi][ni], false, false);
    __syncthreads();
  }

  // ---- epilogue: C layout row = vgpr + 8*(lane>=16), col = lane%16 ----
#pragma unroll
  for (int mi = 0; mi < 2; ++mi)
#pragma unroll
    for (int ni = 0; ni < 4; ++ni)
#pragma unroll
      for (int r = 0; r < 8; ++r) {
        int grow = blockM + wm * 32 + mi * 16 + r + 8 * mh;
        int gcol = blockN + wn * 64 + ni * 16 + mn;
        float v = c[mi][ni][r];
        if (bias) v += bias[gcol];
        if (act) v = gelu_exact(v);
        size_t idx = (size_t)grow * N + gcol;
        if (Cf) Cf[idx] = v;
        if (Cb) Cb[idx] = f2bf(v);
      }
}

// --------------------------- fused attention --------------------------------
// One wave handles one 16-row Q block for one (b,h). Flash-style online
// softmax; P staged through wave-private LDS to convert C-layout -> A-layout.
__global__ __launch_bounds__(128) void attn_kernel(
    const unsigned short* __restrict__ Q, const unsigned short* __restrict__ Km,
    const unsigned short* __restrict__ V, unsigned short* __restrict__ O) {
  __shared__ __align__(16) unsigned short P[4][16 * 32];

  const int lane = threadIdx.x & 31;
  const int wave = threadIdx.x >> 5;
  const int h = blockIdx.y, b = blockIdx.z;
  const int s0 = (blockIdx.x * 4 + wave) * 16;
  const int mn = lane & 15, mh = lane >> 4;

  const unsigned short* qp = Q + ((size_t)b * SD + s0) * DD + h * DK;

  v16bf qa[2];
#pragma unroll
  for (int kh = 0; kh < 2; ++kh) {
    v8u r;
#pragma unroll
    for (int v = 0; v < 8; ++v) {
      int klocal = (v >> 2) * 16 + mh * 8 + (v & 3) * 2;
      r[v] = *(const unsigned*)(qp + (size_t)mn * DD + kh * 32 + klocal);
    }
    qa[kh] = __builtin_bit_cast(v16bf, r);
  }

  v8f o[4] = {};
  float mrun[8], srun[8];
#pragma unroll
  for (int r = 0; r < 8; ++r) { mrun[r] = -3.0e38f; srun[r] = 0.0f; }
  const float sc = 0.125f;              // 1/sqrt(dk=64)
  const float L2E = 1.4426950408889634f;

  for (int j = 0; j < SD; j += 32) {
    // S = Q @ K^T for 32 keys (two 16x16 C tiles)
    v8f s[2] = {};
#pragma unroll
    for (int t = 0; t < 2; ++t)
#pragma unroll
      for (int kh = 0; kh < 2; ++kh) {
        v8u r;
#pragma unroll
        for (int v = 0; v < 8; ++v) {
          int kf = (v >> 2) * 16 + mh * 8 + (v & 3) * 2;
          int key = j + t * 16 + mn;
          r[v] = *(const unsigned*)(Km + ((size_t)b * SD + key) * DD + h * DK +
                                    kh * 32 + kf);
        }
        s[t] = __builtin_amdgcn_wmma_f32_16x16x32_bf16(
            false, qa[kh], false, __builtin_bit_cast(v16bf, r), (short)0, s[t],
            false, false);
      }

    // online softmax (rows live across 16-lane halves: row = r + 8*mh)
#pragma unroll
    for (int r = 0; r < 8; ++r) {
      float x0 = s[0][r] * sc, x1 = s[1][r] * sc;
      float mx = fmaxf(x0, x1);
#pragma unroll
      for (int msk = 1; msk < 16; msk <<= 1) mx = fmaxf(mx, __shfl_xor(mx, msk, 32));
      float mnew = fmaxf(mrun[r], mx);
      float fcorr = __builtin_exp2f((mrun[r] - mnew) * L2E);
      float p0 = __builtin_exp2f((x0 - mnew) * L2E);
      float p1 = __builtin_exp2f((x1 - mnew) * L2E);
      float ps = p0 + p1;
#pragma unroll
      for (int msk = 1; msk < 16; msk <<= 1) ps += __shfl_xor(ps, msk, 32);
      srun[r] = srun[r] * fcorr + ps;
      mrun[r] = mnew;
#pragma unroll
      for (int cc = 0; cc < 4; ++cc) o[cc][r] *= fcorr;
      int prow = r + 8 * mh;
      P[wave][prow * 32 + mn]      = f2bf(p0);
      P[wave][prow * 32 + 16 + mn] = f2bf(p1);
    }
    // wave-private C->A relayout through LDS; explicit CDNA5 DS counter wait
    asm volatile("s_wait_dscnt 0" ::: "memory");

    v8u pr;
#pragma unroll
    for (int v = 0; v < 8; ++v) {
      int klocal = (v >> 2) * 16 + mh * 8 + (v & 3) * 2;
      pr[v] = *(const unsigned*)&P[wave][mn * 32 + klocal];
    }
    v16bf pa = __builtin_bit_cast(v16bf, pr);

    // O += P @ V (dk=64 -> 4 N tiles)
#pragma unroll
    for (int cc = 0; cc < 4; ++cc) {
      v8u rv;
#pragma unroll
      for (int v = 0; v < 8; ++v) {
        int kk = (v >> 2) * 16 + mh * 8 + (v & 3) * 2;  // key pair
        const unsigned short* vp =
            V + ((size_t)b * SD + j + kk) * DD + h * DK + cc * 16 + mn;
        unsigned lo = vp[0], hi = vp[DD];
        rv[v] = lo | (hi << 16);
      }
      o[cc] = __builtin_amdgcn_wmma_f32_16x16x32_bf16(
          false, pa, false, __builtin_bit_cast(v16bf, rv), (short)0, o[cc],
          false, false);
    }
  }

#pragma unroll
  for (int r = 0; r < 8; ++r) {
    float inv = 1.0f / srun[r];
    int row = s0 + r + 8 * mh;
#pragma unroll
    for (int cc = 0; cc < 4; ++cc)
      O[((size_t)b * SD + row) * DD + h * DK + cc * 16 + mn] = f2bf(o[cc][r] * inv);
  }
}

// ------------------------- residual + LayerNorm -----------------------------
__global__ __launch_bounds__(256) void add_ln(
    const float* __restrict__ X, const float* __restrict__ R,
    const float* __restrict__ g, const float* __restrict__ be,
    float* __restrict__ Y) {
  __shared__ float redS[8], redS2[8];
  const size_t row = blockIdx.x;
  const int tid = threadIdx.x;
  float v[4];
  float s = 0.f, s2 = 0.f;
#pragma unroll
  for (int i = 0; i < 4; ++i) {
    int idx = tid + i * 256;
    float t = X[row * DD + idx] + R[row * DD + idx];
    v[i] = t; s += t; s2 += t * t;
  }
#pragma unroll
  for (int msk = 1; msk < 32; msk <<= 1) {
    s  += __shfl_xor(s, msk, 32);
    s2 += __shfl_xor(s2, msk, 32);
  }
  if ((tid & 31) == 0) { redS[tid >> 5] = s; redS2[tid >> 5] = s2; }
  __syncthreads();
  float ts = 0.f, ts2 = 0.f;
#pragma unroll
  for (int i = 0; i < 8; ++i) { ts += redS[i]; ts2 += redS2[i]; }
  float mean = ts * (1.0f / DD);
  float var  = ts2 * (1.0f / DD) - mean * mean;
  float rstd = rsqrtf(var + 1e-3f);
#pragma unroll
  for (int i = 0; i < 4; ++i) {
    int idx = tid + i * 256;
    Y[row * DD + idx] = g[idx] * (v[i] - mean) * rstd + be[idx];
  }
}

// ----------------------------------------------------------------------------
extern "C" void kernel_launch(void* const* d_in, const int* in_sizes, int n_in,
                              void* d_out, int out_size, void* d_ws, size_t ws_size,
                              hipStream_t stream) {
  (void)in_sizes; (void)n_in; (void)out_size; (void)ws_size;
  const float* x   = (const float*)d_in[0];
  const float* WQ  = (const float*)d_in[1];
  const float* WK  = (const float*)d_in[2];
  const float* WV  = (const float*)d_in[3];
  const float* WO  = (const float*)d_in[4];
  const float* K1  = (const float*)d_in[5];
  const float* b1  = (const float*)d_in[6];
  const float* K2  = (const float*)d_in[7];
  const float* b2  = (const float*)d_in[8];
  const float* g1  = (const float*)d_in[9];
  const float* be1 = (const float*)d_in[10];
  const float* g2  = (const float*)d_in[11];
  const float* be2 = (const float*)d_in[12];

  const size_t MROWS = (size_t)BD * SD;              // 4096
  char* w = (char*)d_ws;
  size_t off = 0;
  auto carve = [&](size_t bytes) {
    void* p = w + off;
    off += (bytes + 255) & ~(size_t)255;
    return p;
  };
  const size_t WDD = (size_t)LNUM * DD * DD;         // 6M elems
  const size_t WDF = (size_t)LNUM * DD * FD;         // 24M elems
  unsigned short* wqb = (unsigned short*)carve(WDD * 2);
  unsigned short* wkb = (unsigned short*)carve(WDD * 2);
  unsigned short* wvb = (unsigned short*)carve(WDD * 2);
  unsigned short* wob = (unsigned short*)carve(WDD * 2);
  unsigned short* k1b = (unsigned short*)carve(WDF * 2);
  unsigned short* k2b = (unsigned short*)carve(WDF * 2);
  float*          xf  = (float*)carve(MROWS * DD * 4);
  unsigned short* xb  = (unsigned short*)carve(MROWS * DD * 2);
  unsigned short* qb  = (unsigned short*)carve(MROWS * DD * 2);
  unsigned short* kb  = (unsigned short*)carve(MROWS * DD * 2);
  unsigned short* vb  = (unsigned short*)carve(MROWS * DD * 2);
  unsigned short* ab  = (unsigned short*)carve(MROWS * DD * 2);
  float*          cf  = (float*)carve(MROWS * DD * 4);
  unsigned short* hb  = (unsigned short*)carve(MROWS * FD * 2);
  // total ~250 MB of workspace

  // weight conversion (bf16 + head permutation folded in)
  cvt_qkv  <<<2048, 256, 0, stream>>>(WQ, wqb, WDD);
  cvt_qkv  <<<2048, 256, 0, stream>>>(WK, wkb, WDD);
  cvt_qkv  <<<2048, 256, 0, stream>>>(WV, wvb, WDD);
  cvt_wo   <<<2048, 256, 0, stream>>>(WO, wob, WDD);
  cvt_plain<<<2048, 256, 0, stream>>>(K1, k1b, WDF);
  cvt_plain<<<2048, 256, 0, stream>>>(K2, k2b, WDF);

  (void)hipMemcpyAsync(xf, x, MROWS * DD * sizeof(float), hipMemcpyDeviceToDevice,
                       stream);

  const dim3 gemmBlk(256);
  const dim3 gemmGridD(DD / 128, MROWS / 128);   // N=1024
  const dim3 gemmGridF(FD / 128, MROWS / 128);   // N=4096
  const dim3 attnGrid(SD / 16 / 4, HH, BD);

  for (int l = 0; l < LNUM; ++l) {
    const size_t wOff  = (size_t)l * DD * DD;
    const size_t w1Off = (size_t)l * DD * FD;
    const size_t w2Off = (size_t)l * FD * DD;

    tdm_prefetch<<<1, 32, 0, stream>>>(wqb + wOff);

    cvt_plain<<<2048, 256, 0, stream>>>(xf, xb, MROWS * DD);

    gemm_bf16<<<gemmGridD, gemmBlk, 0, stream>>>(xb, wqb + wOff, (int)MROWS, DD, DD,
                                                 nullptr, 0, nullptr, qb);
    gemm_bf16<<<gemmGridD, gemmBlk, 0, stream>>>(xb, wkb + wOff, (int)MROWS, DD, DD,
                                                 nullptr, 0, nullptr, kb);
    gemm_bf16<<<gemmGridD, gemmBlk, 0, stream>>>(xb, wvb + wOff, (int)MROWS, DD, DD,
                                                 nullptr, 0, nullptr, vb);

    attn_kernel<<<attnGrid, dim3(128), 0, stream>>>(qb, kb, vb, ab);

    gemm_bf16<<<gemmGridD, gemmBlk, 0, stream>>>(ab, wob + wOff, (int)MROWS, DD, DD,
                                                 nullptr, 0, cf, nullptr);
    add_ln<<<(unsigned)MROWS, 256, 0, stream>>>(xf, cf, g1 + (size_t)l * DD,
                                                be1 + (size_t)l * DD, xf);

    cvt_plain<<<2048, 256, 0, stream>>>(xf, xb, MROWS * DD);
    gemm_bf16<<<gemmGridF, gemmBlk, 0, stream>>>(xb, k1b + w1Off, (int)MROWS, FD, DD,
                                                 b1 + (size_t)l * FD, 1, nullptr, hb);
    gemm_bf16<<<gemmGridD, gemmBlk, 0, stream>>>(hb, k2b + w2Off, (int)MROWS, DD, FD,
                                                 b2 + (size_t)l * DD, 0, cf, nullptr);
    add_ln<<<(unsigned)MROWS, 256, 0, stream>>>(xf, cf, g2 + (size_t)l * DD,
                                                be2 + (size_t)l * DD, xf);
  }

  (void)hipMemcpyAsync(d_out, xf, MROWS * DD * sizeof(float),
                       hipMemcpyDeviceToDevice, stream);
}
